// GPT_21062519620130
// MI455X (gfx1250) — matrix operand order repaired and checked by reading, not compiled, not run on hardware
//
#include <hip/hip_runtime.h>
#include <hip/hip_fp16.h>
#include <hip/hip_bf16.h>

// ---- problem constants (match reference) ----
#define B_  2
#define T_  1024
#define C_  768
#define H_  12
#define L_  6
#define D_  64
#define FF_ 3072
#define NROWS (B_*T_)   // 2048

typedef _Float16 half_t;
typedef __attribute__((ext_vector_type(16))) _Float16 v16h;
typedef __attribute__((ext_vector_type(8)))  float    v8f;
typedef unsigned int u32;
typedef u32 u32x4 __attribute__((ext_vector_type(4)));

union Frag { u32x4 q[2]; v16h h; };   // 16 halfs = 2 x 16B

__device__ inline u32 pack_f16x2(float a, float b) {
    union { _Float16 h[2]; u32 u; } x;
    x.h[0] = (_Float16)a; x.h[1] = (_Float16)b;
    return x.u;
}

__device__ inline float gelu_f(float x) {
    const float k = 0.7978845608028654f;
    return 0.5f * x * (1.f + tanhf(k * (x + 0.044715f * x * x * x)));
}

// ---------------- embedding: x = wte[idx] + wpe ----------------
__global__ void embed_kernel(const int* __restrict__ idx, const float* __restrict__ wte,
                             const float* __restrict__ wpe, float* __restrict__ x) {
    int row = blockIdx.x;            // b*T + t
    int tok = idx[row];
    int t = row % T_;
    for (int c = threadIdx.x; c < C_; c += blockDim.x)
        x[row * C_ + c] = wte[(long)tok * C_ + c] + wpe[t * C_ + c];
}

// ---------------- layernorm -> f16 ----------------
__global__ void ln_kernel(const float* __restrict__ x, const float* __restrict__ g,
                          const float* __restrict__ b, half_t* __restrict__ out) {
    __shared__ float s1[256], s2[256];
    int row = blockIdx.x;
    const float* xr = x + row * C_;
    float sum = 0.f, sq = 0.f;
    for (int c = threadIdx.x; c < C_; c += 256) { float v = xr[c]; sum += v; sq += v * v; }
    s1[threadIdx.x] = sum; s2[threadIdx.x] = sq; __syncthreads();
    for (int o = 128; o > 0; o >>= 1) {
        if (threadIdx.x < o) { s1[threadIdx.x] += s1[threadIdx.x + o]; s2[threadIdx.x] += s2[threadIdx.x + o]; }
        __syncthreads();
    }
    float mu = s1[0] * (1.f / C_);
    float var = s2[0] * (1.f / C_) - mu * mu;
    float rstd = rsqrtf(var + 1e-5f);
    for (int c = threadIdx.x; c < C_; c += 256)
        out[row * C_ + c] = (half_t)((xr[c] - mu) * rstd * g[c] + b[c]);
}

// ---- pack fp32 weight [K,N] -> f16 fragment-major tiles ----
// Tile = 32(k) x 16(n). Layout: Wp[((tK*tilesN + tN)*32 + lane)*8 + v]
// lane = (n&15) + 16*((k&31)>=16), v = (k&15)>>1, dword = half2(W[k,n], W[k+1,n]).
// n >= N (padding up to Npad, multiple of 64) packs zeros -> no guards in GEMM loop.
__global__ void pack_w_kernel(const float* __restrict__ W, u32* __restrict__ Wp,
                              int K, int N, int Npad) {
    int id = blockIdx.x * 256 + threadIdx.x;          // output dword index
    int total = (K >> 1) * Npad;
    if (id >= total) return;
    int v    = id & 7;
    int lane = (id >> 3) & 31;
    int tile = id >> 8;
    int tilesN = Npad >> 4;
    int tN = tile % tilesN, tK = tile / tilesN;
    int n = tN * 16 + (lane & 15);
    int k = tK * 32 + (lane >> 4) * 16 + 2 * v;
    float a = 0.f, b = 0.f;
    if (n < N) { a = W[k * N + n]; b = W[(k + 1) * N + n]; }
    Wp[id] = pack_f16x2(a, b);
}

// ---------------- WMMA GEMM: C[M,N] = A[M,K](f16) @ B(tile-packed f16) ----------------
// block = 256 thr = 8 waves; wave tile 32x32; block tile 128x64.
__launch_bounds__(256)
__global__ void wmma_gemm_kernel(const half_t* __restrict__ A, const u32x4* __restrict__ B4,
                                 const float* __restrict__ bias, const float* __restrict__ resid,
                                 float* __restrict__ Cf, half_t* __restrict__ Ch,
                                 int M, int N, int Npad, int K, int do_gelu) {
    int lane = threadIdx.x & 31, wave = threadIdx.x >> 5;
    int half_ = lane >> 4, ln = lane & 15;
    int bm = blockIdx.y * 128 + (wave >> 1) * 32;
    int bn = blockIdx.x * 64 + (wave & 1) * 32;

    const u32x4* A4 = (const u32x4*)A;                 // row stride K/8 uint4
    int a0base = (bm + ln) * (K >> 3) + 2 * half_;
    int a1base = (bm + 16 + ln) * (K >> 3) + 2 * half_;
    int tilesN = Npad >> 4;
    int bstep  = tilesN * 64;                          // uint4 per 32-k step
    int bidx   = (bn >> 4) * 64 + lane * 2;            // tile (kc=0, nt=bn/16), this lane

    v8f acc[2][2] = {};
    for (int k0 = 0; k0 < K; k0 += 32) {
        Frag a0, a1, b0, b1;
        int ko = k0 >> 3;
        a0.q[0] = A4[a0base + ko];  a0.q[1] = A4[a0base + ko + 1];
        a1.q[0] = A4[a1base + ko];  a1.q[1] = A4[a1base + ko + 1];
        b0.q[0] = B4[bidx];         b0.q[1] = B4[bidx + 1];
        b1.q[0] = B4[bidx + 64];    b1.q[1] = B4[bidx + 65];
        __builtin_prefetch((const void*)(B4 + bidx + bstep), 0, 3);
        acc[0][0] = __builtin_amdgcn_wmma_f32_16x16x32_f16(false, a0.h, false, b0.h, (short)0, acc[0][0], false, false);
        acc[0][1] = __builtin_amdgcn_wmma_f32_16x16x32_f16(false, a0.h, false, b1.h, (short)0, acc[0][1], false, false);
        acc[1][0] = __builtin_amdgcn_wmma_f32_16x16x32_f16(false, a1.h, false, b0.h, (short)0, acc[1][0], false, false);
        acc[1][1] = __builtin_amdgcn_wmma_f32_16x16x32_f16(false, a1.h, false, b1.h, (short)0, acc[1][1], false, false);
        bidx += bstep;
    }

    #pragma unroll
    for (int i = 0; i < 2; ++i) {
        #pragma unroll
        for (int j = 0; j < 2; ++j) {
            int n = bn + j * 16 + ln;
            if (n >= N) continue;
            float bv = bias ? bias[n] : 0.f;
            #pragma unroll
            for (int r = 0; r < 8; ++r) {
                int m = bm + i * 16 + half_ * 8 + r;   // C-matrix: vgpr r = row r (+8 upper lanes)
                float v = acc[i][j][r] + bv;
                if (do_gelu) v = gelu_f(v);
                int o = m * N + n;
                if (resid) v += resid[o];
                if (Cf) Cf[o] = v;
                if (Ch) Ch[o] = (half_t)v;
            }
        }
    }
}

// ---- split qkv[2048,3C] f32 into attention-ready f16 layouts ----
// Qf : [B*H][T][D] row-major f16 (A operand of Q@K^T)
// Kt : fragment-major B tiles for S = Q@K^T (k=d: 2 k-chunks, n=s: 64 tiles)
//      idx = (((bh*2 + kc)*64 + st)*32 + lane)*8 + v
// Vt : fragment-major B tiles for O = P@V (k=s: 32 k-chunks, n=d: 4 tiles)
//      idx = (((bh*32 + sc)*4 + dt)*32 + lane)*8 + v
__global__ void qkv_rearrange_kernel(const float* __restrict__ qkv, half_t* __restrict__ Qf,
                                     u32* __restrict__ Kt, u32* __restrict__ Vt) {
    int id = blockIdx.x * 256 + threadIdx.x;
    if (id >= B_ * H_ * T_ * D_) return;
    int d = id % D_;
    int t = (id / D_) % T_;
    int h = (id / (D_ * T_)) % H_;
    int b = id / (D_ * T_ * H_);
    int bh = b * H_ + h;
    const float* rowp = qkv + (b * T_ + t) * (3 * C_);
    int col = h * D_ + d;
    Qf[(bh * T_ + t) * D_ + d] = (half_t)rowp[col];
    if ((d & 1) == 0) {   // K: B[k=d][n=t]
        int kc = d >> 5, st = t >> 4;
        int lane = (t & 15) + 16 * ((d >> 4) & 1);
        int v = (d & 15) >> 1;
        Kt[(((bh * 2 + kc) * 64 + st) * 32 + lane) * 8 + v] =
            pack_f16x2(rowp[C_ + col], rowp[C_ + col + 1]);
    }
    if ((t & 1) == 0) {   // V: B[k=t][n=d]
        const float* rowp1 = rowp + 3 * C_;
        int sc = t >> 5, dt = d >> 4;
        int lane = (d & 15) + 16 * ((t >> 4) & 1);
        int v = (t & 15) >> 1;
        Vt[(((bh * 32 + sc) * 4 + dt) * 32 + lane) * 8 + v] =
            pack_f16x2(rowp[2 * C_ + col], rowp1[2 * C_ + col]);
    }
}

// ---------------- causal flash attention, 1 wave per (bh, 16-row tile) ----------------
__launch_bounds__(32)
__global__ void attn_kernel(const half_t* __restrict__ Qf, const u32x4* __restrict__ Kt4,
                            const u32x4* __restrict__ Vt4, half_t* __restrict__ Oout) {
    __shared__ half_t Pl[16 * 32];           // P chunk [16 rows][32 s], A-fragment staging
    int lane = threadIdx.x;
    int half_ = lane >> 4, ln = lane & 15;
    int t0 = blockIdx.x * 16;
    int bh = blockIdx.y, b = bh / H_, h = bh % H_;

    const u32x4* Q4 = (const u32x4*)(Qf + bh * T_ * D_);   // row stride D/8 = 8 uint4
    const u32x4* Pl4 = (const u32x4*)Pl;

    // Q A-fragments for k=0..31 (aq0) and k=32..63 (aq1), reused every chunk
    int rowQ = t0 + ln;
    Frag aq0, aq1;
    aq0.q[0] = Q4[rowQ * 8 + 2 * half_];        aq0.q[1] = Q4[rowQ * 8 + 2 * half_ + 1];
    aq1.q[0] = Q4[rowQ * 8 + 4 + 2 * half_];    aq1.q[1] = Q4[rowQ * 8 + 4 + 2 * half_ + 1];

    float m[8], lsum[8];
    #pragma unroll
    for (int r = 0; r < 8; ++r) { m[r] = -INFINITY; lsum[r] = 0.f; }
    v8f oacc[4] = {};

    int nchunk = t0 / 32 + 1;
    for (int c = 0; c < nchunk; ++c) {
        int s0 = c * 32;
        v8f sacc[2] = {};
        #pragma unroll
        for (int st = 0; st < 2; ++st) {
            Frag bk0, bk1;
            int stIdx = (s0 >> 4) + st;
            int i0 = ((bh * 2 + 0) * 64 + stIdx) * 64 + lane * 2;   // kc=0
            int i1 = ((bh * 2 + 1) * 64 + stIdx) * 64 + lane * 2;   // kc=1
            bk0.q[0] = Kt4[i0]; bk0.q[1] = Kt4[i0 + 1];
            bk1.q[0] = Kt4[i1]; bk1.q[1] = Kt4[i1 + 1];
            sacc[st] = __builtin_amdgcn_wmma_f32_16x16x32_f16(false, aq0.h, false, bk0.h, (short)0, sacc[st], false, false);
            sacc[st] = __builtin_amdgcn_wmma_f32_16x16x32_f16(false, aq1.h, false, bk1.h, (short)0, sacc[st], false, false);
        }
        bool last = (c == nchunk - 1);
        #pragma unroll
        for (int st = 0; st < 2; ++st)
            #pragma unroll
            for (int r = 0; r < 8; ++r) {
                float sv = sacc[st][r] * 0.125f;                    // D^-0.5
                if (last) {
                    int tr = t0 + r + half_ * 8;
                    int sc = s0 + st * 16 + ln;
                    if (sc > tr) sv = -INFINITY;
                }
                sacc[st][r] = sv;
            }
        // online softmax: row reductions across the 16 lanes of each half
        #pragma unroll
        for (int r = 0; r < 8; ++r) {
            float mx = fmaxf(sacc[0][r], sacc[1][r]);
            for (int off = 1; off < 16; off <<= 1) mx = fmaxf(mx, __shfl_xor(mx, off, 32));
            float mnew = fmaxf(m[r], mx);
            float scl = __expf(m[r] - mnew);
            m[r] = mnew;
            float p0 = __expf(sacc[0][r] - mnew);
            float p1 = __expf(sacc[1][r] - mnew);
            sacc[0][r] = p0; sacc[1][r] = p1;
            float rs = p0 + p1;
            for (int off = 1; off < 16; off <<= 1) rs += __shfl_xor(rs, off, 32);
            lsum[r] = lsum[r] * scl + rs;
            #pragma unroll
            for (int j = 0; j < 4; ++j) oacc[j][r] *= scl;
        }
        // P -> LDS (f16), reload as A-fragment
        #pragma unroll
        for (int st = 0; st < 2; ++st)
            #pragma unroll
            for (int r = 0; r < 8; ++r)
                Pl[(r + half_ * 8) * 32 + st * 16 + ln] = (half_t)sacc[st][r];
        __syncthreads();
        Frag ap;
        ap.q[0] = Pl4[ln * 4 + 2 * half_];
        ap.q[1] = Pl4[ln * 4 + 2 * half_ + 1];
        #pragma unroll
        for (int j = 0; j < 4; ++j) {
            Frag bv;
            int iv = ((bh * 32 + c) * 4 + j) * 64 + lane * 2;
            bv.q[0] = Vt4[iv]; bv.q[1] = Vt4[iv + 1];
            oacc[j] = __builtin_amdgcn_wmma_f32_16x16x32_f16(false, ap.h, false, bv.h, (short)0, oacc[j], false, false);
        }
        __syncthreads();
    }
    // write O / l  -> attn buffer [B*T][C] f16 (A operand of proj GEMM)
    #pragma unroll
    for (int r = 0; r < 8; ++r) {
        float inv = 1.f / lsum[r];
        int trow = t0 + r + half_ * 8;
        int orow = (b * T_ + trow) * C_ + h * D_;
        #pragma unroll
        for (int j = 0; j < 4; ++j)
            Oout[orow + j * 16 + ln] = (half_t)(oacc[j][r] * inv);
    }
}

// ---------------- host orchestration ----------------
extern "C" void kernel_launch(void* const* d_in, const int* in_sizes, int n_in,
                              void* d_out, int out_size, void* d_ws, size_t ws_size,
                              hipStream_t stream) {
    (void)out_size; (void)ws_size;
    const int*   idx    = (const int*)d_in[0];
    const float* wte    = (const float*)d_in[1];
    const float* wpe    = (const float*)d_in[2];
    const float* ln1_g  = (const float*)d_in[3];
    const float* ln1_b  = (const float*)d_in[4];
    const float* wqkv   = (const float*)d_in[5];
    const float* bqkv   = (const float*)d_in[6];
    const float* wproj  = (const float*)d_in[7];
    const float* bproj  = (const float*)d_in[8];
    const float* ln2_g  = (const float*)d_in[9];
    const float* ln2_b  = (const float*)d_in[10];
    const float* wfc    = (const float*)d_in[11];
    const float* bfc    = (const float*)d_in[12];
    const float* wfc2   = (const float*)d_in[13];
    const float* bfc2   = (const float*)d_in[14];
    const float* lnf_g  = (const float*)d_in[15];
    const float* lnf_b  = (const float*)d_in[16];
    const float* w_head = (const float*)d_in[17];
    float* out = (float*)d_out;
    int V = (n_in > 17 && in_sizes[17] > 0) ? in_sizes[17] / C_ : 50257;
    int Vpad = (V + 63) & ~63;

    char* ws = (char*)d_ws;
    size_t off = 0;
    auto alloc = [&](size_t bytes) -> void* {
        void* p = ws + off;
        off += (bytes + 255) & ~(size_t)255;
        return p;
    };
    float*  x     = (float*) alloc((size_t)NROWS * C_ * 4);
    half_t* hbuf  = (half_t*)alloc((size_t)NROWS * C_ * 2);
    float*  qkvb  = (float*) alloc((size_t)NROWS * 3 * C_ * 4);
    half_t* Qf    = (half_t*)alloc((size_t)B_ * H_ * T_ * D_ * 2);
    u32*    Kt    = (u32*)   alloc((size_t)B_ * H_ * (D_ / 2) * T_ * 4);
    u32*    Vt    = (u32*)   alloc((size_t)B_ * H_ * (T_ / 2) * D_ * 4);
    half_t* attnb = (half_t*)alloc((size_t)NROWS * C_ * 2);
    half_t* mlpb  = (half_t*)alloc((size_t)NROWS * FF_ * 2);
    u32*    wbuf  = (u32*)   alloc((size_t)(C_ / 2) * (size_t)Vpad * 4);   // largest packed weight

    auto pack = [&](const float* W, int K, int N, int Npad) {
        int tot = (K / 2) * Npad;
        pack_w_kernel<<<(tot + 255) / 256, 256, 0, stream>>>(W, wbuf, K, N, Npad);
    };
    auto gemm = [&](const half_t* A, const float* bias, const float* resid,
                    float* Cf, half_t* Ch, int M, int N, int Npad, int K, int do_gelu) {
        dim3 g((N + 63) / 64, M / 128);
        wmma_gemm_kernel<<<g, 256, 0, stream>>>(A, (const u32x4*)wbuf, bias, resid,
                                                Cf, Ch, M, N, Npad, K, do_gelu);
    };

    embed_kernel<<<NROWS, 256, 0, stream>>>(idx, wte, wpe, x);

    for (int l = 0; l < L_; ++l) {
        // LN1 -> h (f16)
        ln_kernel<<<NROWS, 256, 0, stream>>>(x, ln1_g + (size_t)l * C_, ln1_b + (size_t)l * C_, hbuf);
        // QKV
        pack(wqkv + (size_t)l * C_ * 3 * C_, C_, 3 * C_, 3 * C_);
        gemm(hbuf, bqkv + (size_t)l * 3 * C_, nullptr, qkvb, nullptr, NROWS, 3 * C_, 3 * C_, C_, 0);
        qkv_rearrange_kernel<<<(B_ * H_ * T_ * D_ + 255) / 256, 256, 0, stream>>>(qkvb, Qf, Kt, Vt);
        // attention
        attn_kernel<<<dim3(T_ / 16, B_ * H_), 32, 0, stream>>>(Qf, (const u32x4*)Kt, (const u32x4*)Vt, attnb);
        // proj + residual
        pack(wproj + (size_t)l * C_ * C_, C_, C_, C_);
        gemm(attnb, bproj + (size_t)l * C_, x, x, nullptr, NROWS, C_, C_, C_, 0);
        // LN2 -> h
        ln_kernel<<<NROWS, 256, 0, stream>>>(x, ln2_g + (size_t)l * C_, ln2_b + (size_t)l * C_, hbuf);
        // FC + GELU -> mlp (f16)
        pack(wfc + (size_t)l * C_ * FF_, C_, FF_, FF_);
        gemm(hbuf, bfc + (size_t)l * FF_, nullptr, nullptr, mlpb, NROWS, FF_, FF_, C_, 1);
        // FC2 + residual
        pack(wfc2 + (size_t)l * FF_ * C_, FF_, C_, C_);
        gemm(mlpb, bfc2 + (size_t)l * C_, x, x, nullptr, NROWS, C_, C_, FF_, 0);
    }

    // final LN + head
    ln_kernel<<<NROWS, 256, 0, stream>>>(x, lnf_g, lnf_b, hbuf);
    pack(w_head, C_, V, Vpad);
    gemm(hbuf, nullptr, nullptr, out, nullptr, NROWS, V, Vpad, C_, 0);
}
